// Correlation_7464653161015
// MI455X (gfx1250) — matrix operand order repaired and checked by reading, not compiled, not run on hardware
//
#include <hip/hip_runtime.h>
#include <stdint.h>

typedef float  v8f   __attribute__((ext_vector_type(8)));
typedef __bf16 v16bf __attribute__((ext_vector_type(16)));

#define B_   16
#define C_   256
#define H_   48
#define W_   64
#define HW   (H_ * W_)
#define ND   21              // displacements per axis (-20..20 step 2)
#define KT   64              // channels staged through LDS per iteration
#define KP   (KT / 2)        // packed K-pair words per row per stage
#define RS   36              // LDS row stride in words (KP=32 + pad) : 16B-aligned rows
#define PLANE (W_ * RS)      // 2304 words per bf16 plane
#define BANDSZ (ND * W_)     // 21 x 64 extracted band

union Frag { uint4 q[2]; v16bf v; };

// round-to-nearest-even fp32 -> bf16 (top 16 bits), integer path (no __bf16 arith)
__device__ __forceinline__ uint32_t bf16_rne(float a) {
    uint32_t u = __float_as_uint(a);
    return (u + 0x7FFFu + ((u >> 16) & 1u)) >> 16;
}

__device__ __forceinline__ void split2(float a0, float a1, uint32_t& hi, uint32_t& lo) {
    uint32_t h0 = bf16_rne(a0), h1 = bf16_rne(a1);
    float r0 = a0 - __uint_as_float(h0 << 16);
    float r1 = a1 - __uint_as_float(h1 << 16);
    hi = h0 | (h1 << 16);
    lo = bf16_rne(r0) | (bf16_rne(r1) << 16);
}

#define WMMA_BF16(A, Bm, Cacc) \
    __builtin_amdgcn_wmma_f32_16x16x32_bf16(false, (A), false, (Bm), (short)0, (Cacc), false, false)

__global__ __launch_bounds__(128)
void corr_wmma_bf16x3_kernel(const float* __restrict__ in1,
                             const float* __restrict__ in2,
                             float* __restrict__ out) {
    __shared__ uint32_t smem[4 * PLANE + BANDSZ];
    uint32_t* s1h = smem;                 // input1 hi plane, [x][kpair] packed
    uint32_t* s1l = smem + PLANE;         // input1 lo plane
    uint32_t* s2h = smem + 2 * PLANE;     // input2 hi plane
    uint32_t* s2l = smem + 3 * PLANE;     // input2 lo plane
    float*    band = (float*)(smem + 4 * PLANE);

    const int tid = threadIdx.x;
    const int oyi = blockIdx.x;           // 0..20
    const int y   = blockIdx.y;           // 0..47
    const int b   = blockIdx.z;           // 0..15
    const int y2  = y + 2 * oyi - 20;

    // out[b, oyi*21 + oxi, y, x]
    const long obase = (((long)b * (ND * ND) + (long)oyi * ND) * H_ + y) * W_;

    if (y2 < 0 || y2 >= H_) {             // whole slice is zero padding
        for (int i = tid; i < BANDSZ; i += 128) {
            int oxi = i >> 6, x = i & 63;
            __builtin_nontemporal_store(0.0f, &out[obase + (long)oxi * HW + x]);
        }
        return;
    }

    const float* g1 = in1 + (((long)b * C_) * H_ + y)  * (long)W_;
    const float* g2 = in2 + (((long)b * C_) * H_ + y2) * (long)W_;

    for (int i = tid; i < BANDSZ; i += 128) band[i] = 0.0f;  // padding zeros

    // ---- wave/lane geometry ----
    const int w    = tid >> 5;            // wave id -> M-tile m0 = 16w
    const int l    = tid & 31;
    const int lm   = l & 15;
    const int m0   = w << 4;
    const int asel = (l >= 16) ? 4 : 0;   // A: lanes 16-31 hold K 8-15 / 24-31
    const int bsel = (l >= 16) ? 4 : 0;   // B: lanes 16-31 hold K 8-15 / 24-31
    const int arow = (m0 + lm) * RS;

    // corner tiles can never reach |x2 - x| <= 20 (min gap 33): skip them
    const bool use0 = (m0 <= 35);         // N-tile 0 unused for m0 = 48
    const bool use3 = (48 - m0 <= 35);    // N-tile 3 unused for m0 = 0

    v8f acc0 = {}, acc1 = {}, acc2 = {}, acc3 = {};

    for (int ks = 0; ks < C_; ks += KT) {
        // ---- stage + split KT channels of both rows into packed bf16 LDS planes ----
        for (int i = tid; i < KP * W_; i += 128) {
            int x = i & 63, cp = i >> 6;                  // kpair index 0..31
            long g = (long)(ks + 2 * cp) * HW + x;
            uint32_t h, lo;
            split2(g1[g], g1[g + HW], h, lo);
            s1h[x * RS + cp] = h;  s1l[x * RS + cp] = lo;
            split2(g2[g], g2[g + HW], h, lo);
            s2h[x * RS + cp] = h;  s2l[x * RS + cp] = lo;
        }
        __syncthreads();

        // ---- 2 K-steps of 32 channels: D += Ahi*Bhi + Ahi*Blo + Alo*Bhi ----
#pragma unroll
        for (int kp = 0; kp < KP; kp += 16) {
            Frag Ah, Al;
            Ah.q[0] = *(const uint4*)(s1h + arow + kp + asel);
            Ah.q[1] = *(const uint4*)(s1h + arow + kp + 8 + asel);
            Al.q[0] = *(const uint4*)(s1l + arow + kp + asel);
            Al.q[1] = *(const uint4*)(s1l + arow + kp + 8 + asel);

#define DO_TILE(NT, ACC)                                                     \
            {                                                                \
                Frag Bh, Bl;                                                 \
                int brow = ((NT) * 16 + lm) * RS + kp + bsel;                \
                Bh.q[0] = *(const uint4*)(s2h + brow);                       \
                Bh.q[1] = *(const uint4*)(s2h + brow + 8);                   \
                Bl.q[0] = *(const uint4*)(s2l + brow);                       \
                Bl.q[1] = *(const uint4*)(s2l + brow + 8);                   \
                ACC = WMMA_BF16(Ah.v, Bh.v, ACC);                            \
                ACC = WMMA_BF16(Ah.v, Bl.v, ACC);                            \
                ACC = WMMA_BF16(Al.v, Bh.v, ACC);                            \
            }

            if (use0) DO_TILE(0, acc0)
            DO_TILE(1, acc1)
            DO_TILE(2, acc2)
            if (use3) DO_TILE(3, acc3)
#undef DO_TILE
        }
        __syncthreads();
    }

    // ---- extract the 21 even diagonals in [-20,20] into the LDS band ----
    // C/D layout: v8f elem r -> row M = r (lanes 0-15) or r+8 (lanes 16-31), col N = lm
    const int xr = m0 + ((l >> 4) << 3);
#pragma unroll
    for (int r = 0; r < 8; ++r) {
        int x = xr + r;
        { int d = ( 0 + lm) - x + 20; if (d >= 0 && d <= 40 && !(d & 1)) band[(d >> 1) * W_ + x] = acc0[r]; }
        { int d = (16 + lm) - x + 20; if (d >= 0 && d <= 40 && !(d & 1)) band[(d >> 1) * W_ + x] = acc1[r]; }
        { int d = (32 + lm) - x + 20; if (d >= 0 && d <= 40 && !(d & 1)) band[(d >> 1) * W_ + x] = acc2[r]; }
        { int d = (48 + lm) - x + 20; if (d >= 0 && d <= 40 && !(d & 1)) band[(d >> 1) * W_ + x] = acc3[r]; }
    }
    __syncthreads();

    // ---- coalesced non-temporal stream of the band to global ----
    for (int i = tid; i < BANDSZ; i += 128) {
        int oxi = i >> 6, x = i & 63;
        __builtin_nontemporal_store(band[i], &out[obase + (long)oxi * HW + x]);
    }
}

extern "C" void kernel_launch(void* const* d_in, const int* in_sizes, int n_in,
                              void* d_out, int out_size, void* d_ws, size_t ws_size,
                              hipStream_t stream) {
    (void)in_sizes; (void)n_in; (void)out_size; (void)d_ws; (void)ws_size;
    const float* in1 = (const float*)d_in[0];
    const float* in2 = (const float*)d_in[1];
    float* out = (float*)d_out;
    dim3 grid(ND, H_, B_);   // (oy index, y, batch) -> 16128 workgroups
    corr_wmma_bf16x3_kernel<<<grid, 128, 0, stream>>>(in1, in2, out);
}
